// Head_2774548873659
// MI455X (gfx1250) — matrix operand order repaired and checked by reading, not compiled, not run on hardware
//
#include <hip/hip_runtime.h>
#include <hip/hip_bf16.h>

#define B_SZ 512
#define T_SZ 256
#define C_SZ 384
#define H_SZ 64

typedef __attribute__((ext_vector_type(16))) _Float16 v16h;
typedef __attribute__((ext_vector_type(8)))  _Float16 v8h;
typedef __attribute__((ext_vector_type(8)))  float    v8f;
typedef __attribute__((ext_vector_type(4)))  float    v4f;

// ---------------- LDS layout (units: _Float16 elements unless noted) ---------
#define QK_STRIDE 80        // 160B rows: 32B-aligned frags, bank spread
#define VT_STRIDE 272       // 544B rows
#define SS_STRIDE 264       // 528B rows
#define WT_STRIDE 392       // 784B rows: lane-bank offset 4n -> conflict free
#define SQ_OFF    0
#define SK_OFF    (T_SZ * QK_STRIDE)                    // 20480
#define SVT_OFF   (SK_OFF + T_SZ * QK_STRIDE)           // 40960
#define SWT_OFF   (SVT_OFF + H_SZ * VT_STRIDE)          // 58368 (phase-1 only)
#define SS_OFF    SWT_OFF                               // phase-2 alias over Wt
#define SSUM_OFF  (SS_OFF + 16 * 16 * SS_STRIDE)        // 125952 (f16 units)
#define SMEM_BYTES ((SWT_OFF + 3 * H_SZ * WT_STRIDE) * 2)   // 267264 bytes

// Convert Wq/Wk/Wv ([C,H] fp32) -> Wt ([3][H][C] f16, transposed) in d_ws.
__global__ void wt_prep_kernel(const float* __restrict__ Wq,
                               const float* __restrict__ Wk,
                               const float* __restrict__ Wv,
                               _Float16* __restrict__ Wt) {
    int i = blockIdx.x * 256 + threadIdx.x;
    if (i >= 3 * H_SZ * C_SZ) return;
    int m = i / (H_SZ * C_SZ);
    int r = i % (H_SZ * C_SZ);
    int h = r / C_SZ;
    int c = r % C_SZ;
    const float* W = (m == 0) ? Wq : (m == 1) ? Wk : Wv;
    Wt[i] = (_Float16)W[c * H_SZ + h];
}

__global__ __launch_bounds__(512, 1)
void attn_fused_kernel(const float* __restrict__ x,
                       const _Float16* __restrict__ Wt,
                       float* __restrict__ out) {
    extern __shared__ char smem_raw[];
    _Float16* sm   = (_Float16*)smem_raw;
    _Float16* sQ   = sm + SQ_OFF;
    _Float16* sK   = sm + SK_OFF;
    _Float16* sVt  = sm + SVT_OFF;
    _Float16* sWt  = sm + SWT_OFF;        // phase 1 only
    _Float16* sS0  = sm + SS_OFF;         // phase 2 alias
    float*    sSum = (float*)(smem_raw + SSUM_OFF * 2);

    const int b    = blockIdx.x;
    const int tid  = threadIdx.x;
    const int wave = tid >> 5;            // 0..15
    const int lane = tid & 31;
    const int lhi  = lane >> 4;
    const int llo  = lane & 15;

    const v8f vzero = {};

    // ---- Phase 0: stage Wt (3*64 rows x 384 f16) into padded LDS ----
    for (int c = tid; c < 192 * 48; c += 512) {          // 48 x v8h chunks/row
        int rowi = c / 48;
        int k    = (c % 48) * 8;
        *(v8h*)(sWt + rowi * WT_STRIDE + k) =
            *(const v8h*)(Wt + (size_t)rowi * C_SZ + k);
    }
    __syncthreads();

    // ================= Phase 1: Q = xWq, K = xWk, V = xWv ====================
    // wave w owns M-tile w (16 rows of x / Q / K / V)
    const float* xb = x + (size_t)b * T_SZ * C_SZ;
    {
        const int mtile = wave;
        v8f acc[12];
#pragma unroll
        for (int j = 0; j < 12; ++j) acc[j] = vzero;

        const float* xrow = xb + (size_t)(mtile * 16 + llo) * C_SZ + 8 * lhi;
        // software pipeline: x chunk for kc+1 loads while kc's WMMAs run
        v4f f0 = *(const v4f*)(xrow);
        v4f f1 = *(const v4f*)(xrow + 4);
        v4f f2 = *(const v4f*)(xrow + 16);
        v4f f3 = *(const v4f*)(xrow + 20);
        for (int kc = 0; kc < 12; ++kc) {
            const int kn = (kc < 11) ? kc + 1 : 11;      // clamped prefetch
            const float* xn = xrow + kn * 32;
            v4f n0 = *(const v4f*)(xn);
            v4f n1 = *(const v4f*)(xn + 4);
            v4f n2 = *(const v4f*)(xn + 16);
            v4f n3 = *(const v4f*)(xn + 20);

            v16h a;
#pragma unroll
            for (int i = 0; i < 4; ++i) {
                a[i]      = (_Float16)f0[i];
                a[4 + i]  = (_Float16)f1[i];
                a[8 + i]  = (_Float16)f2[i];
                a[12 + i] = (_Float16)f3[i];
            }
            // B fragments from LDS Wt: row = output column h, two 8-f16 runs (k = 8h.., 16+8h..)
            const _Float16* wb0 = sWt + llo * WT_STRIDE + kc * 32 + 8 * lhi;
#pragma unroll
            for (int m3 = 0; m3 < 3; ++m3) {
#pragma unroll
                for (int nt = 0; nt < 4; ++nt) {
                    const _Float16* wb = wb0 + (m3 * H_SZ + nt * 16) * WT_STRIDE;
                    v8h blo = *(const v8h*)(wb);
                    v8h bhi = *(const v8h*)(wb + 16);
                    v16h bfrag;
#pragma unroll
                    for (int i = 0; i < 8; ++i) { bfrag[i] = blo[i]; bfrag[8 + i] = bhi[i]; }
                    acc[m3 * 4 + nt] = __builtin_amdgcn_wmma_f32_16x16x32_f16(
                        false, a, false, bfrag, (short)0, acc[m3 * 4 + nt], false, false);
                }
            }
            f0 = n0; f1 = n1; f2 = n2; f3 = n3;
        }
        // Store to LDS as f16: Q,K row-major; V transposed (Vt[h][t], packed)
#pragma unroll
        for (int m3 = 0; m3 < 3; ++m3) {
#pragma unroll
            for (int nt = 0; nt < 4; ++nt) {
                v8f c = acc[m3 * 4 + nt];
                int h = nt * 16 + llo;
                if (m3 == 2) {
                    v8h pk;
#pragma unroll
                    for (int r = 0; r < 8; ++r) pk[r] = (_Float16)c[r];
                    *(v8h*)(sVt + h * VT_STRIDE + mtile * 16 + 8 * lhi) = pk;
                } else {
#pragma unroll
                    for (int r = 0; r < 8; ++r) {
                        int t = mtile * 16 + 8 * lhi + r;
                        _Float16 hv = (_Float16)c[r];
                        if (m3 == 0) sQ[t * QK_STRIDE + h] = hv;
                        else         sK[t * QK_STRIDE + h] = hv;
                    }
                }
            }
        }
    }
    __syncthreads();   // phase boundary: Wt region becomes score buffers

    // ================= Phase 2: causal attention =============================
    // wave w owns query row-tile rt = w
    _Float16* sS     = sS0 + wave * 16 * SS_STRIDE;   // per-wave 16x256 score buf
    float*    rowSum = sSum + wave * 16;
    const int rt = wave;

    // ---- S = (Q @ K^T) * scale, lower-triangle column tiles only ----
    for (int ct = 0; ct <= rt; ++ct) {
        v8f acc = vzero;
#pragma unroll
        for (int hc = 0; hc < 2; ++hc) {
            const _Float16* qp = sQ + (rt * 16 + llo) * QK_STRIDE + hc * 32 + 16 * lhi;
            v8h alo = *(const v8h*)(qp);
            v8h ahi = *(const v8h*)(qp + 8);
            v16h a;
#pragma unroll
            for (int i = 0; i < 8; ++i) { a[i] = alo[i]; a[8 + i] = ahi[i]; }
            const _Float16* kp = sK + (ct * 16 + llo) * QK_STRIDE + hc * 32 + 16 * lhi;
            v16h bfrag = *(const v16h*)kp;
            acc = __builtin_amdgcn_wmma_f32_16x16x32_f16(
                false, a, false, bfrag, (short)0, acc, false, false);
        }
#pragma unroll
        for (int r = 0; r < 8; ++r)
            sS[(8 * lhi + r) * SS_STRIDE + ct * 16 + llo] = (_Float16)(acc[r] * 0.125f);
    }
    asm volatile("s_wait_dscnt 0" ::: "memory");

    // ---- row softmax: unnormalized exp kept in sS, masked cols -> 0 ----
    {
        const int r16  = lane >> 1;       // 2 lanes per row
        const int half = lane & 1;        // 128 cols each
        const int row  = rt * 16 + r16;
        _Float16* srow = sS + r16 * SS_STRIDE + half * 128;
        float mx = -3.0e38f;
        for (int j = 0; j < 16; ++j) {
            v8h chunk = *(const v8h*)(srow + j * 8);
#pragma unroll
            for (int e = 0; e < 8; ++e) {
                int col = half * 128 + j * 8 + e;
                float v = (col <= row) ? (float)chunk[e] : -3.0e38f;
                mx = fmaxf(mx, v);
            }
        }
        mx = fmaxf(mx, __shfl_xor(mx, 1, 32));
        float sum = 0.f;
        for (int j = 0; j < 16; ++j) {
            v8h chunk = *(const v8h*)(srow + j * 8);
            v8h outc;
#pragma unroll
            for (int e = 0; e < 8; ++e) {
                int col = half * 128 + j * 8 + e;
                float ev = 0.f;
                if (col <= row) ev = __expf((float)chunk[e] - mx);
                sum += ev;
                outc[e] = (_Float16)ev;
            }
            *(v8h*)(srow + j * 8) = outc;   // zeros masked region too
        }
        sum += __shfl_xor(sum, 1, 32);
        if (half == 0) rowSum[r16] = sum;
    }
    asm volatile("s_wait_dscnt 0" ::: "memory");

    // ---- out = (P @ V) / rowsum ; only K-steps that can be nonzero ----
    const int ksteps = (rt + 2) >> 1;     // ceil((rt+1)/2) chunks of 32
#pragma unroll
    for (int nt = 0; nt < 4; ++nt) {
        v8f acc = vzero;
        for (int kc = 0; kc < ksteps; ++kc) {
            const _Float16* pp = sS + llo * SS_STRIDE + kc * 32 + 16 * lhi;
            v8h alo = *(const v8h*)(pp);
            v8h ahi = *(const v8h*)(pp + 8);
            v16h a;
#pragma unroll
            for (int i = 0; i < 8; ++i) { a[i] = alo[i]; a[8 + i] = ahi[i]; }
            const _Float16* vp = sVt + (nt * 16 + llo) * VT_STRIDE + kc * 32 + 16 * lhi;
            v16h bfrag = *(const v16h*)vp;
            acc = __builtin_amdgcn_wmma_f32_16x16x32_f16(
                false, a, false, bfrag, (short)0, acc, false, false);
        }
#pragma unroll
        for (int r = 0; r < 8; ++r) {
            float inv = 1.0f / rowSum[8 * lhi + r];
            int t = rt * 16 + 8 * lhi + r;
            out[((size_t)b * T_SZ + t) * H_SZ + nt * 16 + llo] = acc[r] * inv;
        }
    }
}

extern "C" void kernel_launch(void* const* d_in, const int* in_sizes, int n_in,
                              void* d_out, int out_size, void* d_ws, size_t ws_size,
                              hipStream_t stream) {
    (void)in_sizes; (void)n_in; (void)out_size; (void)ws_size;
    const float* x  = (const float*)d_in[0];
    const float* Wq = (const float*)d_in[1];
    const float* Wk = (const float*)d_in[2];
    const float* Wv = (const float*)d_in[3];
    float* out = (float*)d_out;
    _Float16* Wt = (_Float16*)d_ws;   // 3*64*384 f16 = 147456 bytes

    const int wt_elems = 3 * H_SZ * C_SZ;
    wt_prep_kernel<<<(wt_elems + 255) / 256, 256, 0, stream>>>(Wq, Wk, Wv, Wt);
    attn_fused_kernel<<<B_SZ, 512, SMEM_BYTES, stream>>>(x, Wt, out);
}
